// GNN_atom_bond_13322988552794
// MI455X (gfx1250) — matrix operand order repaired and checked by reading, compiled, not run on hardware
//
#include <hip/hip_runtime.h>

// ---------------------------------------------------------------------------
// Types for CDNA5 WMMA (wave32, gfx1250)
// ---------------------------------------------------------------------------
typedef __bf16 bf16_t;
typedef __attribute__((ext_vector_type(16))) __bf16 v16bf;
typedef __attribute__((ext_vector_type(8)))  __bf16 v8bf;
typedef __attribute__((ext_vector_type(8)))  float   v8f;

union AFrag { v16bf v; v8bf h[2]; };

// Problem constants (match reference)
#define BB      128
#define N_ATOMS 128
#define N_BONDS 256
#define DD      256
#define KNB     6
#define RADIUS  3
#define MA      (BB * N_ATOMS)   // 16384 atom rows
#define MB      (BB * N_BONDS)   // 32768 bond rows
#define KA0     64               // 40 padded to 64
#define KB0     32               // 14 padded to 32

// ---------------------------------------------------------------------------
// Pack a [Kin, 256] fp32 weight matrix into WMMA B-fragment order (bf16),
// zero padding K up to Kpad (multiple of 32).
// Layout: for k-tile kt, n-tile nt (16 cols), lane l, element j:
//   offset = ((kt*16 + nt)*32 + l)*16 + j   holds  W[kt*32 + (l>>4)*16 + j][nt*16 + (l&15)]
// in contiguous k order (k = 16*(l>>4) + j, same as the A loads), so the GEMM loads
// one contiguous v16bf (32B) per lane per fragment.
// ---------------------------------------------------------------------------
__global__ __launch_bounds__(256) void k_pack_w(const float* __restrict__ W,
                                                bf16_t* __restrict__ Wp,
                                                int Kin, int Kpad) {
  int e = blockIdx.x * 256 + threadIdx.x;
  if (e >= Kpad * 256) return;
  int j    = e & 15;
  int lane = (e >> 4) & 31;
  int nt   = (e >> 9) & 15;
  int kt   = e >> 13;
  int n = nt * 16 + (lane & 15);
  int k = kt * 32 + (lane >> 4) * 16 + j;
  Wp[e] = (k < Kin) ? (bf16_t)W[(size_t)k * 256 + n] : (bf16_t)0.0f;
}

// fp32 [M, Din] -> bf16 [M, 1<<kshift] zero padded
__global__ __launch_bounds__(256) void k_cvt_pad(const float* __restrict__ X,
                                                 bf16_t* __restrict__ Y,
                                                 int Din, int kshift, long total) {
  long i = (long)blockIdx.x * 256 + threadIdx.x;
  if (i >= total) return;
  int  kpad = 1 << kshift;
  long r = i >> kshift;
  int  c = (int)(i & (kpad - 1));
  Y[i] = (c < Din) ? (bf16_t)X[r * Din + c] : (bf16_t)0.0f;
}

// ---------------------------------------------------------------------------
// Fused GEMM:  out = act(A[M,K]@W[K,256] + bias) (+ resid)
// A: row-major bf16, K multiple of 32.  Wp: packed B-fragments (above).
// Block = 256 threads = 8 waves; wave w covers rows [(w&3)*32,+32) and
// cols [(w>>2)*64,+64) of a 128x128 block tile -> 8 v_wmma_f32_16x16x32_bf16
// per 32-wide k-step per wave.
// ---------------------------------------------------------------------------
__global__ __launch_bounds__(256) void k_gemm_n256(
    const bf16_t* __restrict__ A, const bf16_t* __restrict__ Wp,
    const float* __restrict__ bias, const float* __restrict__ resid,
    float* __restrict__ outF, bf16_t* __restrict__ outB,
    int K, int do_relu)
{
  const int tid  = threadIdx.x;
  const int wave = tid >> 5;
  const int lane = tid & 31;
  const int l15  = lane & 15;
  const int lhi  = lane >> 4;
  const int rowBase = blockIdx.x * 128 + (wave & 3) * 32;
  const int colBase = blockIdx.y * 128 + (wave >> 2) * 64;
  const int ntBase  = colBase >> 4;

  v8f acc[2][4];
#pragma unroll
  for (int mi = 0; mi < 2; ++mi)
#pragma unroll
    for (int c = 0; c < 4; ++c) acc[mi][c] = (v8f){};

  // Per-lane A addresses for the 16x32 A fragment:
  // lane<16: row=l15, K = kb..kb+15 (contiguous, matching packed B)
  // lane>=16: same rows, kb shifted by 16.
  const bf16_t* aPtr0 = A + (size_t)(rowBase + l15) * K + lhi * 16;
  const bf16_t* aPtr1 = aPtr0 + (size_t)16 * K;

  const int ksteps = K >> 5;
  for (int kt = 0; kt < ksteps; ++kt) {
    AFrag a0, a1;
    a0.h[0] = *(const v8bf*)(aPtr0 + kt * 32);
    a0.h[1] = *(const v8bf*)(aPtr0 + kt * 32 + 8);
    a1.h[0] = *(const v8bf*)(aPtr1 + kt * 32);
    a1.h[1] = *(const v8bf*)(aPtr1 + kt * 32 + 8);
#pragma unroll
    for (int c = 0; c < 4; ++c) {
      const v16bf bfrag =
          *(const v16bf*)(Wp + ((((size_t)kt * 16) + ntBase + c) * 32 + lane) * 16);
      acc[0][c] = __builtin_amdgcn_wmma_f32_16x16x32_bf16(
          false, a0.v, false, bfrag, (short)0, acc[0][c], false, false);
      acc[1][c] = __builtin_amdgcn_wmma_f32_16x16x32_bf16(
          false, a1.v, false, bfrag, (short)0, acc[1][c], false, false);
    }
  }

  // Epilogue. C/D layout: v8f element j -> row = base + lhi*8 + j, col = base + l15.
#pragma unroll
  for (int mi = 0; mi < 2; ++mi) {
#pragma unroll
    for (int c = 0; c < 4; ++c) {
      const int col = colBase + c * 16 + l15;
      const float bv = bias[col];
#pragma unroll
      for (int j = 0; j < 8; ++j) {
        const int row = rowBase + mi * 16 + lhi * 8 + j;
        const size_t o = (size_t)row * 256 + col;
        float v = acc[mi][c][j] + bv;
        if (do_relu) v = fmaxf(v, 0.0f);
        if (resid) v += resid[o];
        if (outF) outF[o] = v;
        if (outB) outB[o] = (bf16_t)v;
      }
    }
  }
}

// ---------------------------------------------------------------------------
// Neighborhood gather + context + (1+eps)*state, emitting bf16 "pre" for the
// following MLP GEMM.  One block per (batch,row), one thread per channel.
// ---------------------------------------------------------------------------
__global__ __launch_bounds__(256) void k_gather_pre(
    const float* __restrict__ nb_self, const float* __restrict__ nb_other,
    const int* __restrict__ idx_self, const int* __restrict__ idx_other,
    const float* __restrict__ state, const float* __restrict__ eps, int eps_i,
    bf16_t* __restrict__ pre_out, int selfShift, int otherCount)
{
  const int row   = blockIdx.x;
  const int d     = threadIdx.x;
  const int nSelf = 1 << selfShift;
  const int b     = row >> selfShift;
  const size_t selfBase  = (size_t)b * nSelf;
  const size_t otherBase = (size_t)b * otherCount;
  const int* is = idx_self  + (size_t)row * KNB;
  const int* io = idx_other + (size_t)row * KNB;

  float ctx = 0.0f;
#pragma unroll
  for (int k = 0; k < KNB; ++k) {
    int ia = is[k];
    int ib = io[k];
    const float att = (ia < nSelf) ? 1.0f : 0.0f;          // attend mask
    ia = (ia < 0) ? 0 : ((ia >= nSelf)      ? nSelf - 1      : ia);
    ib = (ib < 0) ? 0 : ((ib >= otherCount) ? otherCount - 1 : ib);
    const float vs = nb_self [(selfBase  + ia) * 256 + d];
    const float vo = nb_other[(otherBase + ib) * 256 + d];
    ctx += att * (vs + vo);
  }
  const float e   = eps[eps_i];
  const float pre = (1.0f + e) * state[(size_t)row * 256 + d] + ctx;
  pre_out[(size_t)row * 256 + d] = (bf16_t)pre;
}

// op_mol_atom[b,d] = sum_n atom_f[b,n,d] * mask[b,n]
__global__ __launch_bounds__(256) void k_mol(const float* __restrict__ atom_f,
                                             const float* __restrict__ mask,
                                             float* __restrict__ out) {
  const int b = blockIdx.x;
  const int d = threadIdx.x;
  float s = 0.0f;
  for (int n = 0; n < N_ATOMS; ++n)
    s += atom_f[((size_t)b * N_ATOMS + n) * 256 + d] * mask[b * N_ATOMS + n];
  out[(size_t)b * 256 + d] = s;
}

// ---------------------------------------------------------------------------
// Host-side orchestration
// ---------------------------------------------------------------------------
static inline size_t bump(size_t& off, size_t bytes) {
  size_t p = off;
  off = (off + bytes + 255) & ~(size_t)255;
  return p;
}

extern "C" void kernel_launch(void* const* d_in, const int* in_sizes, int n_in,
                              void* d_out, int out_size, void* d_ws, size_t ws_size,
                              hipStream_t stream) {
  (void)in_sizes; (void)n_in; (void)out_size; (void)ws_size;

  // ---- inputs (setup_inputs() dict order; params flattened in insertion order)
  const float* x_atoms   = (const float*)d_in[0];
  const float* x_bonds   = (const float*)d_in[1];
  const float* atom_mask = (const float*)d_in[2];
  // d_in[3] bond_mask unused by reference outputs
  const int* idx_aa = (const int*)d_in[4];
  const int* idx_ab = (const int*)d_in[5];
  const int* idx_bb = (const int*)d_in[6];
  const int* idx_ba = (const int*)d_in[7];

  const float* W_a0 = (const float*)d_in[8];   const float* b_a0 = (const float*)d_in[9];
  const float* W_a1 = (const float*)d_in[10];  const float* b_a1 = (const float*)d_in[11];
  const float* W_b0 = (const float*)d_in[12];  const float* b_b0 = (const float*)d_in[13];
  const float* W_b1 = (const float*)d_in[14];  const float* b_b1 = (const float*)d_in[15];

  const float *W_anb[RADIUS], *b_anb[RADIUS], *W_bnb[RADIUS], *b_bnb[RADIUS];
  const float *W_aml[RADIUS], *b_aml[RADIUS], *W_bml[RADIUS], *b_bml[RADIUS];
  int p = 16;
  for (int i = 0; i < RADIUS; ++i) { W_anb[i] = (const float*)d_in[p++]; b_anb[i] = (const float*)d_in[p++]; }
  for (int i = 0; i < RADIUS; ++i) { W_bnb[i] = (const float*)d_in[p++]; b_bnb[i] = (const float*)d_in[p++]; }
  for (int i = 0; i < RADIUS; ++i) { W_aml[i] = (const float*)d_in[p++]; b_aml[i] = (const float*)d_in[p++]; }
  for (int i = 0; i < RADIUS; ++i) { W_bml[i] = (const float*)d_in[p++]; b_bml[i] = (const float*)d_in[p++]; }
  const float* eps = (const float*)d_in[p];

  // ---- workspace carve
  char* ws = (char*)d_ws;
  size_t off = 0;
  bf16_t* wp_a0 = (bf16_t*)(ws + bump(off, (size_t)KA0 * 256 * 2));
  bf16_t* wp_a1 = (bf16_t*)(ws + bump(off, (size_t)256 * 256 * 2));
  bf16_t* wp_b0 = (bf16_t*)(ws + bump(off, (size_t)KB0 * 256 * 2));
  bf16_t* wp_b1 = (bf16_t*)(ws + bump(off, (size_t)256 * 256 * 2));
  bf16_t *wp_anb[RADIUS], *wp_bnb[RADIUS], *wp_aml[RADIUS], *wp_bml[RADIUS];
  for (int i = 0; i < RADIUS; ++i) wp_anb[i] = (bf16_t*)(ws + bump(off, (size_t)256 * 256 * 2));
  for (int i = 0; i < RADIUS; ++i) wp_bnb[i] = (bf16_t*)(ws + bump(off, (size_t)256 * 256 * 2));
  for (int i = 0; i < RADIUS; ++i) wp_aml[i] = (bf16_t*)(ws + bump(off, (size_t)256 * 256 * 2));
  for (int i = 0; i < RADIUS; ++i) wp_bml[i] = (bf16_t*)(ws + bump(off, (size_t)256 * 256 * 2));

  bf16_t* a_in   = (bf16_t*)(ws + bump(off, (size_t)MA * KA0 * 2));
  bf16_t* b_in   = (bf16_t*)(ws + bump(off, (size_t)MB * KB0 * 2));
  bf16_t* tmpA   = (bf16_t*)(ws + bump(off, (size_t)MA * 256 * 2));   // hidden / pre (atoms)
  bf16_t* tmpB   = (bf16_t*)(ws + bump(off, (size_t)MB * 256 * 2));   // hidden / pre (bonds)
  float*  af32   = (float*) (ws + bump(off, (size_t)MA * 256 * 4));   // atom_f state
  bf16_t* abf    = (bf16_t*)(ws + bump(off, (size_t)MA * 256 * 2));
  float*  bf32   = (float*) (ws + bump(off, (size_t)MB * 256 * 4));   // bond_f state
  bf16_t* bbf    = (bf16_t*)(ws + bump(off, (size_t)MB * 256 * 2));
  float*  anb    = (float*) (ws + bump(off, (size_t)MA * 256 * 4));   // atom_nb_t
  float*  bnb    = (float*) (ws + bump(off, (size_t)MB * 256 * 4));   // bond_nb_t

  float* outAtom = (float*)d_out;
  float* outBond = outAtom + (size_t)MA * 256;
  float* outMol  = outBond + (size_t)MB * 256;

  const dim3 blk(256);
  auto packGrid = [](int kpad) { return dim3((kpad * 256 + 255) / 256); };

  // ---- pack weights (bf16, WMMA-B fragment order)
  k_pack_w<<<packGrid(KA0), blk, 0, stream>>>(W_a0, wp_a0, 40, KA0);
  k_pack_w<<<packGrid(256), blk, 0, stream>>>(W_a1, wp_a1, 256, 256);
  k_pack_w<<<packGrid(KB0), blk, 0, stream>>>(W_b0, wp_b0, 14, KB0);
  k_pack_w<<<packGrid(256), blk, 0, stream>>>(W_b1, wp_b1, 256, 256);
  for (int i = 0; i < RADIUS; ++i) {
    k_pack_w<<<packGrid(256), blk, 0, stream>>>(W_anb[i], wp_anb[i], 256, 256);
    k_pack_w<<<packGrid(256), blk, 0, stream>>>(W_bnb[i], wp_bnb[i], 256, 256);
    k_pack_w<<<packGrid(256), blk, 0, stream>>>(W_aml[i], wp_aml[i], 256, 256);
    k_pack_w<<<packGrid(256), blk, 0, stream>>>(W_bml[i], wp_bml[i], 256, 256);
  }

  // ---- convert/pad inputs to bf16
  {
    long ta = (long)MA * KA0, tb = (long)MB * KB0;
    k_cvt_pad<<<dim3((unsigned)((ta + 255) / 256)), blk, 0, stream>>>(x_atoms, a_in, 40, 6, ta);
    k_cvt_pad<<<dim3((unsigned)((tb + 255) / 256)), blk, 0, stream>>>(x_bonds, b_in, 14, 5, tb);
  }

  const dim3 gA(MA / 128, 2), gB(MB / 128, 2);

  // ---- PreGNN MLPs (Linear+ReLU x2)
  k_gemm_n256<<<gA, blk, 0, stream>>>(a_in, wp_a0, b_a0, nullptr, nullptr, tmpA, KA0, 1);
  k_gemm_n256<<<gA, blk, 0, stream>>>(tmpA, wp_a1, b_a1, nullptr, af32, abf, 256, 1);
  k_gemm_n256<<<gB, blk, 0, stream>>>(b_in, wp_b0, b_b0, nullptr, nullptr, tmpB, KB0, 1);
  k_gemm_n256<<<gB, blk, 0, stream>>>(tmpB, wp_b1, b_b1, nullptr, bf32, bbf, 256, 1);

  // ---- message-passing radius loop
  for (int i = 0; i < RADIUS; ++i) {
    // neighbor transforms (linear, no relu)
    k_gemm_n256<<<gA, blk, 0, stream>>>(abf, wp_anb[i], b_anb[i], nullptr, anb, nullptr, 256, 0);
    k_gemm_n256<<<gB, blk, 0, stream>>>(bbf, wp_bnb[i], b_bnb[i], nullptr, bnb, nullptr, 256, 0);

    // gather + context + (1+eps)*state -> bf16 pre-activations
    k_gather_pre<<<dim3(MA), blk, 0, stream>>>(anb, bnb, idx_aa, idx_ab, af32, eps, i,
                                               tmpA, 7, N_BONDS);
    k_gather_pre<<<dim3(MB), blk, 0, stream>>>(bnb, anb, idx_bb, idx_ba, bf32, eps, i,
                                               tmpB, 8, N_ATOMS);

    // update MLP: relu(pre @ W + b) + state  (residual); last radius writes d_out
    const bool last = (i == RADIUS - 1);
    k_gemm_n256<<<gA, blk, 0, stream>>>(tmpA, wp_aml[i], b_aml[i], af32,
                                        last ? outAtom : af32, last ? nullptr : abf, 256, 1);
    k_gemm_n256<<<gB, blk, 0, stream>>>(tmpB, wp_bml[i], b_bml[i], bf32,
                                        last ? outBond : bf32, last ? nullptr : bbf, 256, 1);

    if (i == 0)  // masked per-molecule readout of atom_f after radius 0
      k_mol<<<dim3(BB), blk, 0, stream>>>(af32, atom_mask, outMol);
  }
}
